// SelfAttention_transformer_v3_65395172049598
// MI455X (gfx1250) — matrix operand order repaired and checked by reading, not compiled, not run on hardware
//
#include <hip/hip_runtime.h>
#include <hip/hip_bf16.h>
#include <stdint.h>

// ---------------------------------------------------------------------------
// MI455X (gfx1250, wave32) self-attention pipeline.
// GEMMs: v_wmma_f32_16x16x32_bf16, LDS double-buffered via
// global_load_async_to_lds_b128 (ASYNCcnt) issued through inline asm.
// ---------------------------------------------------------------------------

typedef __attribute__((ext_vector_type(16))) __bf16 v16bf;
typedef __attribute__((ext_vector_type(8)))  float  v8f;

struct Bits32 { uint4 lo; uint4 hi; };   // 32 bytes == one v16bf fragment

__device__ __forceinline__ unsigned short f2bf(float f) {
    unsigned int u = __float_as_uint(f);
    u += 0x7FFFu + ((u >> 16) & 1u);          // round-to-nearest-even
    return (unsigned short)(u >> 16);
}

__device__ __forceinline__ v8f zero8() {
    v8f z;
#pragma unroll
    for (int i = 0; i < 8; ++i) z[i] = 0.0f;
    return z;
}

// Two 16-byte chunks `elemGap` bf16 elements apart (A frag: 16, B frag: 16).
__device__ __forceinline__ v16bf load_frag2(const unsigned short* p, int elemGap) {
    Bits32 s;
    s.lo = *reinterpret_cast<const uint4*>(p);
    s.hi = *reinterpret_cast<const uint4*>(p + elemGap);
    return __builtin_bit_cast(v16bf, s);
}

// ---- CDNA5 async global->LDS copy (ASYNCcnt path) via inline asm -----------
// GLOBAL_LOAD_ASYNC_TO_LDS_B128 (VGLOBAL op 98): VDST = per-lane LDS byte
// offset, VADDR = 64-bit global address, SADDR = off.
__device__ __forceinline__ void cp16_g2l(const unsigned short* g, unsigned short* l) {
    const unsigned lds = (unsigned)(unsigned long long)
        (__attribute__((address_space(3))) const void*)l;    // LDS byte offset
    asm volatile("global_load_async_to_lds_b128 %0, %1, off"
                 :
                 : "v"(lds), "v"((unsigned long long)g)
                 : "memory");
}

__device__ __forceinline__ void async_wait0() {
    asm volatile("s_wait_asynccnt 0" ::: "memory");          // SOPP op 74
}

// ---------------------------------------------------------------------------
// C[M,N] = A[M,K] * Bt[N,K]^T (+bias)
//   A row-major bf16 (lda), Bt N-major with contiguous K (ldb).
// Block: 256 threads = 8 waves arranged 2(M) x 4(N).
// Block tile 128x256, wave tile 64x64 (4x4 WMMA), K-step 32.
// LDS: double-buffered A(128x32) + B(256x32), rows padded to 40 elem (80B).
// mode 0: fp32 row-major; 1: bf16 row-major; 2: bf16 "vt" transposed store
//         out[b*1024*2048 + col*2048 + s], row = b*2048+s.
// ---------------------------------------------------------------------------
#define LDS_PITCH 40                       // 32 + 8 pad (elements)
#define A_ELEMS   (128 * LDS_PITCH)       // 5120
#define B_ELEMS   (256 * LDS_PITCH)       // 10240
#define BUF_ELEMS (A_ELEMS + B_ELEMS)     // 15360 (30720 B / buffer)

__global__ __launch_bounds__(256) void gemm_bf16_wmma(
    const unsigned short* __restrict__ A,  int lda, long long strideA,
    const unsigned short* __restrict__ Bt, int ldb, long long strideB,
    const float* __restrict__ bias,
    float* __restrict__ outF, long long strideOF,
    unsigned short* __restrict__ outB,
    int mode, int M, int N, int Kdim)
{
    __shared__ unsigned short smem[2][BUF_ELEMS];

    A  += (size_t)blockIdx.y * (size_t)strideA;
    Bt += (size_t)blockIdx.y * (size_t)strideB;
    if (outF) outF += (size_t)blockIdx.y * (size_t)strideOF;

    const int nbCount = N >> 8;                         // block tiles along N
    const int mb = (int)blockIdx.x / nbCount;
    const int nb = (int)blockIdx.x - mb * nbCount;
    const int m0 = mb << 7;                             // block row origin
    const int n0 = nb << 8;                             // block col origin

    const int waveId = (int)(threadIdx.x >> 5);
    const int lane   = (int)(threadIdx.x & 31);
    const int wm = (waveId >> 2) << 6;                  // 0 / 64   (M within block)
    const int wn = (waveId & 3)  << 6;                  // 0..192   (N within block)
    const int hl = lane >> 4;
    const int ll = lane & 15;

    unsigned short* const As0 = smem[0];
    unsigned short* const Bs0 = smem[0] + A_ELEMS;
    unsigned short* const As1 = smem[1];
    unsigned short* const Bs1 = smem[1] + A_ELEMS;

    // Cooperative staging of one K-step: 512 A-units + 1024 B-units of 16B,
    // 1536 = 6 * 256 -> 6 copies per thread, fully uniform control flow.
    const int t = (int)threadIdx.x;
    auto stage = [&](int kk, int buf) {
        unsigned short* const Asb = buf ? As1 : As0;
        unsigned short* const Bsb = buf ? Bs1 : Bs0;
#pragma unroll
        for (int u = 0; u < 6; ++u) {
            const int id = t + u * 256;
            if (id < 512) {
                const int row = id >> 2, c = id & 3;
                cp16_g2l(A + (size_t)(m0 + row) * lda + kk + (c << 3),
                         Asb + row * LDS_PITCH + (c << 3));
            } else {
                const int v = id - 512;
                const int row = v >> 2, c = v & 3;
                cp16_g2l(Bt + (size_t)(n0 + row) * ldb + kk + (c << 3),
                         Bsb + row * LDS_PITCH + (c << 3));
            }
        }
    };

    v8f acc[4][4];
#pragma unroll
    for (int i = 0; i < 4; ++i)
#pragma unroll
        for (int j = 0; j < 4; ++j) acc[i][j] = zero8();

    stage(0, 0);
    int cur = 0;

    for (int k0 = 0; k0 < Kdim; k0 += 32) {
        async_wait0();            // my async copies into buf[cur] have landed
        __syncthreads();          // everyone's copies landed / prev reads done

        if (k0 + 32 < Kdim) stage(k0 + 32, cur ^ 1);

        const unsigned short* const Asb = cur ? As1 : As0;
        const unsigned short* const Bsb = cur ? Bs1 : Bs0;

        v16bf af[4], bfr[4];
#pragma unroll
        for (int i = 0; i < 4; ++i)
            af[i] = load_frag2(Asb + (wm + i * 16 + ll) * LDS_PITCH + (hl << 3), 16);
#pragma unroll
        for (int j = 0; j < 4; ++j)
            bfr[j] = load_frag2(Bsb + (wn + j * 16 + ll) * LDS_PITCH + (hl << 3), 16);

#pragma unroll
        for (int i = 0; i < 4; ++i)
#pragma unroll
            for (int j = 0; j < 4; ++j)
                acc[i][j] = __builtin_amdgcn_wmma_f32_16x16x32_bf16(
                    false, af[i], false, bfr[j], (short)0, acc[i][j], false, false);

        cur ^= 1;
    }

    // Epilogue. C/D layout: VGPR r -> (M = r + 8*hl, N = ll) per 16x16 tile.
#pragma unroll
    for (int i = 0; i < 4; ++i) {
        const int rowb = m0 + wm + i * 16 + (hl << 3);
#pragma unroll
        for (int j = 0; j < 4; ++j) {
            const int col  = n0 + wn + j * 16 + ll;
            const float bv = bias ? bias[col] : 0.0f;
#pragma unroll
            for (int r = 0; r < 8; ++r) {
                const float v  = acc[i][j][r] + bv;
                const int row  = rowb + r;
                if (mode == 0)      outF[(size_t)row * N + col] = v;
                else if (mode == 1) outB[(size_t)row * N + col] = f2bf(v);
                else {
                    const int bb = row >> 11, ssx = row & 2047;
                    outB[((size_t)bb * 1024 + col) * 2048 + ssx] = f2bf(v);
                }
            }
        }
    }
}

// ---------------------------------------------------------------------------
// xb[b][s][j] = bf16( j<1024 ? inp[b][s][j] : PE[s][j-1024] )
// ---------------------------------------------------------------------------
__global__ __launch_bounds__(256) void build_xb(const float* __restrict__ inp,
                                                unsigned short* __restrict__ xb)
{
    const size_t i  = (size_t)blockIdx.x * 256 + threadIdx.x;   // B*S*2048 total
    const int    j  = (int)(i & 2047);
    const size_t bs = i >> 11;              // b*S + s
    const int    s  = (int)(bs & 2047);
    float val;
    if (j < 1024) {
        val = inp[(bs << 10) + j];
    } else if (s == 0) {
        val = 0.0f;                          // PE row 0 is zeroed
    } else {
        const int   j2    = j - 1024;
        const float e     = (float)(2 * (j2 >> 1)) * (1.0f / 1024.0f);
        const float denom = __powf(10000.0f, e);
        const float ang   = (float)s / denom;
        val = (j2 & 1) ? __cosf(ang) : __sinf(ang);
    }
    xb[i] = f2bf(val);
}

// Wt[n][k] = bf16(W[k][n]);  W is [2048 x 1024] row-major.
__global__ __launch_bounds__(256) void pack_w(const float* __restrict__ W,
                                              unsigned short* __restrict__ Wt)
{
    const int idx = (int)(blockIdx.x * 256 + threadIdx.x);      // 2048*1024
    const int k   = idx >> 10;
    const int n   = idx & 1023;
    Wt[(size_t)n * 2048 + k] = f2bf(W[(size_t)k * 1024 + n]);
}

// ---------------------------------------------------------------------------
// Column softmax over the QUERY axis, post-scaled by 1/sqrt(1024)=1/32.
// Block = 64 columns x 4 q-lanes; coalesced; online max/sum.
// ---------------------------------------------------------------------------
__global__ __launch_bounds__(256) void col_softmax(const float* __restrict__ scores,
                                                   unsigned short* __restrict__ attn)
{
    const int c      = (int)(threadIdx.x & 63);
    const int qq     = (int)(threadIdx.x >> 6);     // 0..3
    const int colIdx = (int)blockIdx.x * 64 + c;    // over B*S columns
    const int b      = colIdx >> 11;
    const int n      = colIdx & 2047;
    const float* base = scores + (size_t)b * 2048 * 2048 + n;

    float m = -3.4e38f, ssum = 0.0f;
    for (int q = qq; q < 2048; q += 4) {
        const float x  = base[(size_t)q * 2048];
        const float nm = fmaxf(m, x);
        ssum = ssum * __expf(m - nm) + __expf(x - nm);
        m = nm;
    }
    __shared__ float sm[256], ss[256];
    sm[threadIdx.x] = m; ss[threadIdx.x] = ssum;
    __syncthreads();
    if (qq == 0) {
        float M0 = sm[c], S0 = ss[c];
#pragma unroll
        for (int tt = 1; tt < 4; ++tt) {
            const float mt = sm[tt * 64 + c], st = ss[tt * 64 + c];
            const float nm = fmaxf(M0, mt);
            S0 = S0 * __expf(M0 - nm) + st * __expf(mt - nm);
            M0 = nm;
        }
        sm[c] = M0; ss[c] = S0;
    }
    __syncthreads();
    const float Mf  = sm[c];
    const float inv = 1.0f / (ss[c] * 32.0f);       // includes /sqrt(dk)
    unsigned short* ab = attn + (size_t)b * 2048 * 2048 + n;
    for (int q = qq; q < 2048; q += 4)
        ab[(size_t)q * 2048] = f2bf(__expf(base[(size_t)q * 2048] - Mf) * inv);
}

// Per-row LayerNorm stats over sa rows of length 1024.
__global__ __launch_bounds__(256) void row_stats(const float* __restrict__ sa,
                                                 float* __restrict__ mu,
                                                 float* __restrict__ rstd)
{
    const int row = (int)blockIdx.x;
    const float* p = sa + (size_t)row * 1024;
    float s1 = 0.0f, s2 = 0.0f;
    for (int j = (int)threadIdx.x; j < 1024; j += 256) {
        const float v = p[j];
        s1 += v; s2 += v * v;
    }
    __shared__ float r1[256], r2[256];
    r1[threadIdx.x] = s1; r2[threadIdx.x] = s2;
    __syncthreads();
    for (int off = 128; off > 0; off >>= 1) {
        if ((int)threadIdx.x < off) {
            r1[threadIdx.x] += r1[threadIdx.x + off];
            r2[threadIdx.x] += r2[threadIdx.x + off];
        }
        __syncthreads();
    }
    if (threadIdx.x == 0) {
        const float mean = r1[0] * (1.0f / 1024.0f);
        const float var  = r2[0] * (1.0f / 1024.0f) - mean * mean;
        mu[row]   = mean;
        rstd[row] = rsqrtf(var + 1e-5f);
    }
}

// Partial sums over 128-query chunks:  sum_q (sa - mu) * rstd   per (b, d).
__global__ __launch_bounds__(256) void ctx_partial(const float* __restrict__ sa,
                                                   const float* __restrict__ mu,
                                                   const float* __restrict__ rstd,
                                                   float* __restrict__ partial)
{
    const int idx = (int)(blockIdx.x * 256 + threadIdx.x);   // over B*1024
    const int b   = idx >> 10;
    const int dd  = idx & 1023;
    const int q0  = (int)blockIdx.y * 128;
    const float* p  = sa   + ((size_t)b * 2048 + q0) * 1024 + dd;
    const float* pm = mu   + (size_t)b * 2048 + q0;
    const float* pr = rstd + (size_t)b * 2048 + q0;
    float acc = 0.0f;
#pragma unroll 4
    for (int q = 0; q < 128; ++q)
        acc += (p[(size_t)q * 1024] - pm[q]) * pr[q];
    partial[(size_t)blockIdx.y * 8192 + idx] = acc;
}

__global__ __launch_bounds__(256) void ctx_final(const float* __restrict__ partial,
                                                 const float* __restrict__ gamma,
                                                 const float* __restrict__ beta,
                                                 float* __restrict__ ctx)
{
    const int idx = (int)(blockIdx.x * 256 + threadIdx.x);   // over B*1024
    const int dd  = idx & 1023;
    float s = 0.0f;
#pragma unroll
    for (int tt = 0; tt < 16; ++tt) s += partial[(size_t)tt * 8192 + idx];
    ctx[idx] = gamma[dd] * (s * (1.0f / 2048.0f)) + beta[dd];
}

// ---------------------------------------------------------------------------
extern "C" void kernel_launch(void* const* d_in, const int* in_sizes, int n_in,
                              void* d_out, int out_size, void* d_ws, size_t ws_size,
                              hipStream_t stream)
{
    (void)in_sizes; (void)n_in; (void)out_size; (void)ws_size;

    const float* inp   = (const float*)d_in[0];
    const float* Wq    = (const float*)d_in[1];
    const float* bq    = (const float*)d_in[2];
    const float* Wk    = (const float*)d_in[3];
    const float* bk    = (const float*)d_in[4];
    const float* Wv    = (const float*)d_in[5];
    const float* bv    = (const float*)d_in[6];
    const float* gamma = (const float*)d_in[7];
    const float* beta  = (const float*)d_in[8];
    // d_in[9] = lens : unused by the reference computation.

    constexpr int B = 8, S = 2048, D = 1024, K2 = 2048;

    // Workspace layout (bytes, all 256-aligned). attn aliases xb (dead then).
    char* ws = (char*)d_ws;
    unsigned short* xb   = (unsigned short*)(ws);               //  67,108,864
    unsigned short* wt   = (unsigned short*)(ws +  67108864);   //  12,582,912 (3 x 4MiB)
    unsigned short* qb   = (unsigned short*)(ws +  79691776);   //  33,554,432
    unsigned short* kb   = (unsigned short*)(ws + 113246208);   //  33,554,432
    unsigned short* vt   = (unsigned short*)(ws + 146800640);   //  33,554,432
    float*          sc   = (float*)         (ws + 180355072);   // 134,217,728
    float*          mu   = (float*)         (ws + 314572800);   //      65,536
    float*          rs   = (float*)         (ws + 314638336);   //      65,536
    float*          par  = (float*)         (ws + 314703872);   //     524,288
    unsigned short* attn = xb;                                  // alias

    float* sa  = (float*)d_out;                 // [B,S,D] fp32
    float* ctx = sa + (size_t)B * S * D;        // [B,D]   fp32

    const dim3 blk(256);

    // 1) x = concat(inp, PE) -> bf16
    build_xb<<<(B * S * K2) / 256, blk, 0, stream>>>(inp, xb);
    // 2) transpose+demote weights: Wt[n][k]
    pack_w<<<(K2 * D) / 256, blk, 0, stream>>>(Wq, wt);
    pack_w<<<(K2 * D) / 256, blk, 0, stream>>>(Wk, wt + 2048 * 1024);
    pack_w<<<(K2 * D) / 256, blk, 0, stream>>>(Wv, wt + 2 * 2048 * 1024);

    // 3) QKV projections  [16384 x 2048] x [2048 x 1024]; 128x256 block tiles
    const int gQKV = ((B * S) / 128) * (D / 256);       // 512 blocks
    gemm_bf16_wmma<<<dim3(gQKV, 1), blk, 0, stream>>>(
        xb, K2, 0, wt,                   K2, 0, bq, nullptr, 0, qb, 1, B * S, D, K2);
    gemm_bf16_wmma<<<dim3(gQKV, 1), blk, 0, stream>>>(
        xb, K2, 0, wt + 2048 * 1024,     K2, 0, bk, nullptr, 0, kb, 1, B * S, D, K2);
    gemm_bf16_wmma<<<dim3(gQKV, 1), blk, 0, stream>>>(
        xb, K2, 0, wt + 2 * 2048 * 1024, K2, 0, bv, nullptr, 0, vt, 2, B * S, D, K2);

    // 4) scores[b] = q[b] @ k[b]^T   (k row-major already is K-contiguous Bt)
    const int gS = (S / 128) * (S / 256);               // 128 blocks x 8 batches
    gemm_bf16_wmma<<<dim3(gS, B), blk, 0, stream>>>(
        qb, D, (long long)S * D, kb, D, (long long)S * D,
        nullptr, sc, (long long)S * S, nullptr, 0, S, S, D);

    // 5) softmax over the query axis (columns), /32 folded in
    col_softmax<<<(B * S) / 64, blk, 0, stream>>>(sc, attn);

    // 6) sa[b] = attn[b] @ v[b]  (vt is K-contiguous Bt), fp32 out -> d_out
    const int gA = (S / 128) * (D / 256);               // 64 blocks x 8 batches
    gemm_bf16_wmma<<<dim3(gA, B), blk, 0, stream>>>(
        attn, S, (long long)S * S, vt, S, (long long)D * S,
        nullptr, sa, (long long)S * D, nullptr, 0, S, D, S);

    // 7) LayerNorm stats + context reduction
    row_stats<<<B * S, blk, 0, stream>>>(sa, mu, rs);
    ctx_partial<<<dim3((B * D) / 256, 16), blk, 0, stream>>>(sa, mu, rs, par);
    ctx_final<<<(B * D) / 256, blk, 0, stream>>>(par, gamma, beta, ctx);
}
